// CorrectorV2_13280038879859
// MI455X (gfx1250) — matrix operand order repaired and checked by reading, not compiled, not run on hardware
//
#include <hip/hip_runtime.h>
#include <hip/hip_bf16.h>

// ---------------------------------------------------------------------------
// Fused CorrectorV2 pipeline for MI455X (gfx1250, wave32, WMMA).
//   B = 32768 rows, H = 256.
//   Algebraic facts exploited:
//     * h == c == 0 into both LSTM cells  ->  Whh matmuls are identically 0.
//     * c == 0                            ->  forget gate never used; skip
//       the f-gate quarter of Wih (25% of the LSTM GEMM FLOPs).
//   Heavy GEMMs run as f16 x f16 -> f32 via v_wmma_f32_16x16x32_f16.
//   Wave tile = 32 rows x 16 cols: 6 independent WMMAs per LSTM K-step
//   (accumulator chain distance 6 -> no hazard NOPs), B fragments reused
//   across both row halves.
// ---------------------------------------------------------------------------

#define BROWS 32768
#define HDIM  256

typedef _Float16 v8h  __attribute__((ext_vector_type(8)));
typedef _Float16 v16h __attribute__((ext_vector_type(16)));
typedef float    v8f  __attribute__((ext_vector_type(8)));

__device__ __forceinline__ v8f wmma_f16(v16h a, v16h b, v8f c) {
  // (neg_a, A, neg_b, B, c_mod, C, reuse_a, reuse_b)
  return __builtin_amdgcn_wmma_f32_16x16x32_f16(false, a, false, b, (short)0, c,
                                                false, false);
}

// A fragment, 16x32 f16, row-major source with leading dim `lda` (f16 units).
// Documented layout: lane<16 -> M=lane, K = {0..7, 16..23};
//                    lane>=16 -> M=lane-16, K = {8..15, 24..31}.
__device__ __forceinline__ v16h load_a_frag(const _Float16* A, int lda, int m0,
                                            int k0, int lane) {
  int m  = m0 + (lane & 15);
  int kb = k0 + ((lane >> 4) << 3);   // +8 for upper half-wave
  const _Float16* p = A + (size_t)m * lda + kb;
  v8h lo = *(const v8h*)(p);          // K kb .. kb+7
  v8h hi = *(const v8h*)(p + 16);     // K kb+16 .. kb+23
  return __builtin_shufflevector(lo, hi, 0, 1, 2, 3, 4, 5, 6, 7, 8, 9, 10, 11,
                                 12, 13, 14, 15);
}

// B fragment from pre-swizzled weights: each 32x16 K-N tile is stored as
// [lane][16 f16] contiguous, so a lane load is two contiguous b128s.
__device__ __forceinline__ v16h load_b_frag(const _Float16* Bp, int tileIdx,
                                            int lane) {
  const _Float16* p = Bp + (size_t)tileIdx * 512 + lane * 16;
  v8h lo = *(const v8h*)(p);
  v8h hi = *(const v8h*)(p + 8);
  return __builtin_shufflevector(lo, hi, 0, 1, 2, 3, 4, 5, 6, 7, 8, 9, 10, 11,
                                 12, 13, 14, 15);
}

__device__ __forceinline__ float sigmoidf_(float x) {
  return 1.0f / (1.0f + __expf(-x));
}

// ---------------------------------------------------------------------------
// Kernel 1: per-row geometry preprocess -> data[63] (f16, padded to 64) + w.
// ---------------------------------------------------------------------------
__global__ void pre_kernel(const float* __restrict__ aM,
                           const float* __restrict__ RMB,
                           const float* __restrict__ weight,
                           const int*   __restrict__ use_flag,
                           _Float16* __restrict__ dataA,
                           float* __restrict__ wclip) {
  int b = blockIdx.x * blockDim.x + threadIdx.x;
  if (b >= BROWS) return;

  float a[6][3];
  const float* ap = aM + (size_t)b * 18;
#pragma unroll
  for (int i = 0; i < 6; ++i)
#pragma unroll
    for (int j = 0; j < 3; ++j) a[i][j] = ap[i * 3 + j];

  float Rm[6][9];
  const float* rp = RMB + (size_t)b * 54;
#pragma unroll
  for (int n = 0; n < 6; ++n)
#pragma unroll
    for (int e = 0; e < 9; ++e) Rm[n][e] = rp[n * 9 + e];

  const float* R6 = Rm[5];
  float d[64];
  // gR = -10 * R6[1,:]
#pragma unroll
  for (int k = 0; k < 3; ++k) d[k] = -10.0f * R6[3 + k];
  // acc[i][k] = sum_j (a[i][j]-a[5][j]) * R6[j][k]
  int idx = 3;
#pragma unroll
  for (int i = 0; i < 5; ++i)
#pragma unroll
    for (int k = 0; k < 3; ++k) {
      float s = 0.f;
#pragma unroll
      for (int j = 0; j < 3; ++j) s += (a[i][j] - a[5][j]) * R6[j * 3 + k];
      d[idx++] = s;
    }
  // ori[n][i][k] = sum_j R6[j][i] * Rm[n][j][k]
#pragma unroll
  for (int n = 0; n < 5; ++n)
#pragma unroll
    for (int i = 0; i < 3; ++i)
#pragma unroll
      for (int k = 0; k < 3; ++k) {
        float s = 0.f;
#pragma unroll
        for (int j = 0; j < 3; ++j) s += R6[j * 3 + i] * Rm[n][j * 3 + k];
        d[idx++] = s;
      }
  d[63] = 0.0f;

  _Float16* dp = dataA + (size_t)b * 64;
#pragma unroll
  for (int e = 0; e < 64; ++e) dp[e] = (_Float16)d[e];

  float delta = (*use_flag) ? (1.0f / 90.0f) : (-1.0f / 90.0f);
  float w = weight[b] + delta;
  w = fminf(fmaxf(w, 0.0f), 1.0f);
  wclip[b] = w;
}

// ---------------------------------------------------------------------------
// Kernel 2: pack B = Wsub^T (K x N) into per-lane WMMA fragment layout (f16).
// B[k][n] = W[(rowStart+n)*ldW + k] for k < Kreal, else 0.
// ---------------------------------------------------------------------------
__global__ void pack_b_kernel(const float* __restrict__ W,
                              _Float16* __restrict__ dst, int Kreal, int KP,
                              int N, int rowStart, int ldW) {
  int tid = blockIdx.x * blockDim.x + threadIdx.x;
  int total = KP * N;
  if (tid >= total) return;
  int k = tid % KP;
  int n = tid / KP;
  int kt = k >> 5;
  int nt = n >> 4;
  int lane = (n & 15) + (((k >> 3) & 1) << 4);
  int s = (k & 7) + (((k >> 4) & 1) << 3);
  float v = (k < Kreal) ? W[(size_t)(rowStart + n) * ldW + k] : 0.0f;
  int NT = N >> 4;
  dst[((size_t)(kt * NT + nt) * 32 + lane) * 16 + s] = (_Float16)v;
}

// ---------------------------------------------------------------------------
// Kernel 3: x0 = relu(data @ W1^T + b1).  A: (B,64) f16. Bp: packed (64,256).
// Block: 128 threads = 4 waves; wave -> 32x16 C tile; grid (B/128, 16).
// ---------------------------------------------------------------------------
__global__ void gemm_relu_kernel(const _Float16* __restrict__ A,
                                 const _Float16* __restrict__ Bp,
                                 const float* __restrict__ b1,
                                 _Float16* __restrict__ Out) {
  int lane = threadIdx.x & 31;
  int wave = threadIdx.x >> 5;
  int m0 = blockIdx.x * 128 + wave * 32;
  int nt = blockIdx.y;

  v8f c[2] = {{}, {}};
#pragma unroll
  for (int kt = 0; kt < 2; ++kt) {
    v16h a0 = load_a_frag(A, 64, m0, kt * 32, lane);
    v16h a1 = load_a_frag(A, 64, m0 + 16, kt * 32, lane);
    v16h bf = load_b_frag(Bp, kt * 16 + nt, lane);
    c[0] = wmma_f16(a0, bf, c[0]);
    c[1] = wmma_f16(a1, bf, c[1]);
  }

  int n = nt * 16 + (lane & 15);
  float bias = b1[n];
#pragma unroll
  for (int half = 0; half < 2; ++half) {
    int mBase = m0 + half * 16 + ((lane >> 4) << 3);
#pragma unroll
    for (int r = 0; r < 8; ++r) {
      float v = c[half][r] + bias;
      v = v > 0.0f ? v : 0.0f;
      Out[(size_t)(mBase + r) * HDIM + n] = (_Float16)v;
    }
  }
}

// ---------------------------------------------------------------------------
// Kernel 4: LSTM layer with h=c=0:  h = sig(o)*tanh(sig(i)*tanh(g)).
// Wave tile: 32 rows x 16 cols x 3 gates = 6 accumulators; one shared pair of
// A fragments and 3 shared B fragments per K step -> 6 independent WMMAs
// back-to-back (no hazard NOPs).  A: (B,256) f16; grid (B/128, 16), block 128.
// ---------------------------------------------------------------------------
__global__ void gemm_lstm_kernel(const _Float16* __restrict__ A,
                                 const _Float16* __restrict__ Bi,
                                 const _Float16* __restrict__ Bg,
                                 const _Float16* __restrict__ Bo,
                                 const float* __restrict__ bih,
                                 const float* __restrict__ bhh,
                                 _Float16* __restrict__ Hout) {
  int lane = threadIdx.x & 31;
  int wave = threadIdx.x >> 5;
  int m0 = blockIdx.x * 128 + wave * 32;
  int nt = blockIdx.y;

  v8f ci[2] = {{}, {}}, cg[2] = {{}, {}}, co[2] = {{}, {}};
#pragma unroll
  for (int kt = 0; kt < 8; ++kt) {
    v16h a0 = load_a_frag(A, HDIM, m0, kt * 32, lane);
    v16h a1 = load_a_frag(A, HDIM, m0 + 16, kt * 32, lane);
    v16h fi = load_b_frag(Bi, kt * 16 + nt, lane);
    v16h fg = load_b_frag(Bg, kt * 16 + nt, lane);
    v16h fo = load_b_frag(Bo, kt * 16 + nt, lane);
    if (kt < 7) {
      // Prefetch next K-step A fragments (lowers to global_prefetch_b8).
      const _Float16* pf =
          A + (size_t)(m0 + (lane & 15)) * HDIM + (kt + 1) * 32 +
          ((lane >> 4) << 3);
      __builtin_prefetch(pf, 0, 0);
      __builtin_prefetch(pf + 16 * HDIM, 0, 0);
    }
    ci[0] = wmma_f16(a0, fi, ci[0]);
    ci[1] = wmma_f16(a1, fi, ci[1]);
    cg[0] = wmma_f16(a0, fg, cg[0]);
    cg[1] = wmma_f16(a1, fg, cg[1]);
    co[0] = wmma_f16(a0, fo, co[0]);
    co[1] = wmma_f16(a1, fo, co[1]);
  }

  int j = nt * 16 + (lane & 15);
  float bi = bih[j]       + bhh[j];        // gate i at row offset 0
  float bg = bih[512 + j] + bhh[512 + j];  // gate g at row offset 2H
  float bo = bih[768 + j] + bhh[768 + j];  // gate o at row offset 3H
#pragma unroll
  for (int half = 0; half < 2; ++half) {
    int mBase = m0 + half * 16 + ((lane >> 4) << 3);
#pragma unroll
    for (int r = 0; r < 8; ++r) {
      float iv = sigmoidf_(ci[half][r] + bi);
      float gv = tanhf(cg[half][r] + bg);
      float ov = sigmoidf_(co[half][r] + bo);
      float h = ov * tanhf(iv * gv);
      Hout[(size_t)(mBase + r) * HDIM + j] = (_Float16)h;
    }
  }
}

// ---------------------------------------------------------------------------
// Kernel 5: out = h2 @ W2^T + b2 (N=10, via LDS-staged W2), then angle math
// and rotations -> aMc, wMc, RMBc.
// d_out layout: [aMc (B*18) | wMc (B*18) | RMBc (B*54)] f32.
// ---------------------------------------------------------------------------
__global__ void head_post_kernel(const _Float16* __restrict__ H2,
                                 const float* __restrict__ W2,
                                 const float* __restrict__ b2,
                                 const float* __restrict__ aM,
                                 const float* __restrict__ wM,
                                 const float* __restrict__ RMB,
                                 const float* __restrict__ wclip,
                                 float* __restrict__ out) {
  __shared__ float sW2[10 * HDIM];
  for (int i = threadIdx.x; i < 10 * HDIM; i += blockDim.x) sW2[i] = W2[i];
  __syncthreads();

  int b = blockIdx.x * blockDim.x + threadIdx.x;
  if (b >= BROWS) return;

  float s[10];
#pragma unroll
  for (int n = 0; n < 10; ++n) s[n] = b2[n];
  const _Float16* hp = H2 + (size_t)b * HDIM;
  for (int k = 0; k < HDIM; ++k) {
    float hv = (float)hp[k];
#pragma unroll
    for (int n = 0; n < 10; ++n) s[n] += hv * sW2[n * HDIM + k];
  }

  float w = wclip[b];
  float th[6];
#pragma unroll
  for (int n = 0; n < 5; ++n) {
    float cr = s[n], sr = s[5 + n];
    float nn = sqrtf(cr * cr + sr * sr);
    float ang = atan2f(sr / nn, cr / nn) * w;
    th[n] = -ang;
  }
  th[5] = 0.0f;

  const float* ap = aM + (size_t)b * 18;
  const float* wp = wM + (size_t)b * 18;
  const float* rp = RMB + (size_t)b * 54;
  float* oA = out + (size_t)b * 18;
  float* oW = out + (size_t)BROWS * 18 + (size_t)b * 18;
  float* oR = out + (size_t)BROWS * 36 + (size_t)b * 54;

#pragma unroll
  for (int n = 0; n < 6; ++n) {
    float c_ = __cosf(th[n]);
    float s_ = __sinf(th[n]);
    // R = [[c,0,s],[0,1,0],[-s,0,c]]
    float a0 = ap[n * 3 + 0], a1 = ap[n * 3 + 1], a2 = ap[n * 3 + 2];
    oA[n * 3 + 0] = c_ * a0 + s_ * a2;
    oA[n * 3 + 1] = a1;
    oA[n * 3 + 2] = -s_ * a0 + c_ * a2;
    float w0 = wp[n * 3 + 0], w1 = wp[n * 3 + 1], w2 = wp[n * 3 + 2];
    oW[n * 3 + 0] = c_ * w0 + s_ * w2;
    oW[n * 3 + 1] = w1;
    oW[n * 3 + 2] = -s_ * w0 + c_ * w2;
#pragma unroll
    for (int k = 0; k < 3; ++k) {
      float m0 = rp[n * 9 + 0 + k];
      float m1 = rp[n * 9 + 3 + k];
      float m2 = rp[n * 9 + 6 + k];
      oR[n * 9 + 0 + k] = c_ * m0 + s_ * m2;
      oR[n * 9 + 3 + k] = m1;
      oR[n * 9 + 6 + k] = -s_ * m0 + c_ * m2;
    }
  }
}

// ---------------------------------------------------------------------------
// Launcher.  Workspace budget ~39 MB:
//   dataA 4MB | buf0 16MB | buf1 16MB | wclip 128KB | packed weights ~0.8MB.
// ---------------------------------------------------------------------------
extern "C" void kernel_launch(void* const* d_in, const int* in_sizes, int n_in,
                              void* d_out, int out_size, void* d_ws,
                              size_t ws_size, hipStream_t stream) {
  const float* aM     = (const float*)d_in[0];
  const float* wM     = (const float*)d_in[1];
  const float* RMB    = (const float*)d_in[2];
  const float* weight = (const float*)d_in[3];
  const float* W1     = (const float*)d_in[4];
  const float* b1     = (const float*)d_in[5];
  const float* Wih0   = (const float*)d_in[6];
  // d_in[7] = Whh0: dead (h==0), only biases survive.
  const float* bih0   = (const float*)d_in[8];
  const float* bhh0   = (const float*)d_in[9];
  const float* Wih1   = (const float*)d_in[10];
  // d_in[11] = Whh1: dead.
  const float* bih1   = (const float*)d_in[12];
  const float* bhh1   = (const float*)d_in[13];
  const float* W2     = (const float*)d_in[14];
  const float* b2     = (const float*)d_in[15];
  const int*   uflag  = (const int*)d_in[16];

  char* wsp = (char*)d_ws;
  auto carve = [&](size_t bytes) -> char* {
    char* p = wsp;
    wsp += (bytes + 255) & ~(size_t)255;
    return p;
  };
  _Float16* dataA = (_Float16*)carve((size_t)BROWS * 64 * 2);
  _Float16* buf0  = (_Float16*)carve((size_t)BROWS * HDIM * 2);  // x0, then h2
  _Float16* buf1  = (_Float16*)carve((size_t)BROWS * HDIM * 2);  // h1
  float*    wclip = (float*)carve((size_t)BROWS * 4);
  _Float16* pW1   = (_Float16*)carve((size_t)64 * 256 * 2);
  _Float16* pG[6];
  for (int i = 0; i < 6; ++i) pG[i] = (_Float16*)carve((size_t)256 * 256 * 2);

  // 1. preprocess
  pre_kernel<<<BROWS / 256, 256, 0, stream>>>(aM, RMB, weight, uflag, dataA,
                                              wclip);

  // 2. pack weights into WMMA B-fragment layout
  {
    int tot = 64 * 256;
    pack_b_kernel<<<(tot + 255) / 256, 256, 0, stream>>>(W1, pW1, 63, 64, 256,
                                                         0, 63);
  }
  {
    int tot = 256 * 256;
    int grid = (tot + 255) / 256;
    // layer 0: gates i (rows 0..255), g (512..767), o (768..1023)
    pack_b_kernel<<<grid, 256, 0, stream>>>(Wih0, pG[0], 256, 256, 256, 0, 256);
    pack_b_kernel<<<grid, 256, 0, stream>>>(Wih0, pG[1], 256, 256, 256, 512, 256);
    pack_b_kernel<<<grid, 256, 0, stream>>>(Wih0, pG[2], 256, 256, 256, 768, 256);
    // layer 1
    pack_b_kernel<<<grid, 256, 0, stream>>>(Wih1, pG[3], 256, 256, 256, 0, 256);
    pack_b_kernel<<<grid, 256, 0, stream>>>(Wih1, pG[4], 256, 256, 256, 512, 256);
    pack_b_kernel<<<grid, 256, 0, stream>>>(Wih1, pG[5], 256, 256, 256, 768, 256);
  }

  // 3. x0 = relu(data @ W1^T + b1)
  {
    dim3 grid(BROWS / 128, 16);
    gemm_relu_kernel<<<grid, 128, 0, stream>>>(dataA, pW1, b1, buf0);
  }
  // 4. h1 = lstm(x0), h2 = lstm(h1)
  {
    dim3 grid(BROWS / 128, 16);
    gemm_lstm_kernel<<<grid, 128, 0, stream>>>(buf0, pG[0], pG[1], pG[2], bih0,
                                               bhh0, buf1);
    gemm_lstm_kernel<<<grid, 128, 0, stream>>>(buf1, pG[3], pG[4], pG[5], bih1,
                                               bhh1, buf0);
  }
  // 5. head + postprocess
  head_post_kernel<<<BROWS / 256, 256, 0, stream>>>(buf0, W2, b2, aM, wM, RMB,
                                                    wclip, (float*)d_out);
}
